// EGCN_27273042330404
// MI455X (gfx1250) — matrix operand order repaired and checked by reading, not compiled, not run on hardware
//
#include <hip/hip_runtime.h>
#include <hip/hip_bf16.h>

typedef __attribute__((ext_vector_type(16))) __bf16 v16bf;
typedef __attribute__((ext_vector_type(8)))  float  v8f;
typedef __attribute__((ext_vector_type(8)))  float  f32x8;

#define DIM 256

// ---------------- row L2-normalize: one wave per node ----------------
__global__ void k_normalize(const float* __restrict__ emb,
                            float* __restrict__ xout, int n_nodes) {
    int wave = (blockIdx.x * blockDim.x + threadIdx.x) >> 5;
    int lane = threadIdx.x & 31;
    if (wave >= n_nodes) return;
    const float* row = emb + (size_t)wave * DIM;
    float v[8];
    float ss = 0.f;
#pragma unroll
    for (int i = 0; i < 8; ++i) { v[i] = row[lane + 32 * i]; ss += v[i] * v[i]; }
#pragma unroll
    for (int off = 16; off > 0; off >>= 1) ss += __shfl_xor(ss, off, 32);
    float nrm = sqrtf(ss);
    float scale = 1.0f / fmaxf(nrm, 1e-12f);
    float* out = xout + (size_t)wave * DIM;
#pragma unroll
    for (int i = 0; i < 8; ++i) out[lane + 32 * i] = v[i] * scale;
}

// ---------------- zero fill (fp32) ----------------
__global__ void k_zero(float* __restrict__ p, long long n4) {
    long long i = (long long)blockIdx.x * blockDim.x + threadIdx.x;
    long long stride = (long long)gridDim.x * blockDim.x;
    float4 z = make_float4(0.f, 0.f, 0.f, 0.f);
    for (; i < n4; i += stride) ((float4*)p)[i] = z;
}

// ---------------- edge scatter: one wave per (symmetrized) edge ----------------
// e < E:  src = ei[0][e], dst = ei[1][e];  e >= E: src = ei[1][e-E], dst = ei[0][e-E]
__global__ void k_scatter(const int* __restrict__ ei, const float* __restrict__ w,
                          const float* __restrict__ xin, float* __restrict__ agg,
                          int E) {
    int wave = (blockIdx.x * blockDim.x + threadIdx.x) >> 5;
    int lane = threadIdx.x & 31;
    if (wave >= 2 * E) return;
    int src, dst;
    if (wave < E) { src = ei[wave];            dst = ei[E + wave]; }
    else          { int e = wave - E; src = ei[E + e]; dst = ei[e]; }
    float wt = w[wave];
    const float* xr = xin + (size_t)src * DIM;
    float*       ar = agg + (size_t)dst * DIM;
#pragma unroll
    for (int i = 0; i < 8; ++i) {
        int c = lane + 32 * i;
        unsafeAtomicAdd(&ar[c], xr[c] * wt);   // -> global_atomic_add_f32
    }
}

// ---------------- pack B = [W_l | W_r] row-major into bf16 [256][512] ----------------
__global__ void k_build_bt(const float* __restrict__ Wl, const float* __restrict__ Wr,
                           __bf16* __restrict__ Bt) {
    int idx = blockIdx.x * blockDim.x + threadIdx.x;
    if (idx >= DIM * 2 * DIM) return;
    int j = idx >> 9;        // output column (row of W)
    int k = idx & 511;       // K index (0..255 -> W_l, 256..511 -> W_r)
    float v = (k < DIM) ? Wl[j * DIM + k] : Wr[j * DIM + (k - DIM)];
    Bt[idx] = (__bf16)v;
}

// ---------------- fused GEMM: Out = epilogue( [agg|xprev] @ Bt^T + bias ) ----------------
// A = [N][512] fp32 split as Apart0=agg, Apart1=xprev (K 0..255 / 256..511)
// Bt = [256][512] bf16 row-major (column j of output reads Bt[j][*] contiguously)
// One wave: 16 rows x 64 cols (4 tiles), K-loop 16 x 32 via v_wmma_f32_16x16x32_bf16.
__global__ void k_gemm(const float* __restrict__ Apart0,
                       const float* __restrict__ Apart1,
                       const __bf16* __restrict__ Bt,
                       const float* __restrict__ bias,
                       const float* __restrict__ add0,   // residual x   (or null)
                       const float* __restrict__ add1,   // residual x1  (or null)
                       float* __restrict__ Out,
                       int n_row_tiles, int n_nodes) {
    int wave_in_block = threadIdx.x >> 5;
    int lane = threadIdx.x & 31;
    int row_tile = blockIdx.x * 8 + wave_in_block;
    if (row_tile >= n_row_tiles) return;          // wave-uniform: EXEC all-ones for WMMA
    int col_group = blockIdx.y;                   // 0..3 -> 64 cols each
    int row_base = row_tile * 16;
    int half = lane >> 4;
    int m    = lane & 15;
    int arow = row_base + m;
    if (arow >= n_nodes) arow = n_nodes - 1;      // clamp tail reads (writes guarded)

    v8f acc[4] = {};

#pragma unroll
    for (int ks = 0; ks < 16; ++ks) {
        int kglob = ks * 32;
        const float* Ap = (kglob < DIM) ? Apart0 : Apart1;
        int kloc = kglob & (DIM - 1);
        // A fragment: elements 0..7 -> K = kloc+half*16+e ; 8..15 -> K = kloc+half*16+8+e
        f32x8 f0 = *(const f32x8*)(Ap + (size_t)arow * DIM + kloc + half * 16);
        f32x8 f1 = *(const f32x8*)(Ap + (size_t)arow * DIM + kloc + half * 16 + 8);
        v16bf a;
#pragma unroll
        for (int i = 0; i < 8; ++i) { a[i] = (__bf16)f0[i]; a[8 + i] = (__bf16)f1[i]; }
#pragma unroll
        for (int t = 0; t < 4; ++t) {
            int ncol = col_group * 64 + t * 16 + (lane & 15);
            // B fragment: element e -> K = kglob + half*16 + e, column ncol
            v16bf b = *(const v16bf*)(Bt + (size_t)ncol * (2 * DIM) + kglob + half * 16);
            acc[t] = __builtin_amdgcn_wmma_f32_16x16x32_bf16(
                false, a, false, b, (short)0, acc[t], false, false);
        }
    }

    // epilogue: bias + leaky_relu(0.01) [+ residuals], D layout: VGPR r -> row m=r+8*half
#pragma unroll
    for (int t = 0; t < 4; ++t) {
        int col = col_group * 64 + t * 16 + (lane & 15);
#pragma unroll
        for (int r = 0; r < 8; ++r) {
            int orow = row_base + r + 8 * half;
            if (orow >= n_nodes) continue;
            float v = acc[t][r] + bias[col];
            v = (v > 0.f) ? v : 0.01f * v;
            size_t o = (size_t)orow * DIM + col;
            if (add0) v += add0[o] + add1[o];
            Out[o] = v;
        }
    }
}

extern "C" void kernel_launch(void* const* d_in, const int* in_sizes, int n_in,
                              void* d_out, int out_size, void* d_ws, size_t ws_size,
                              hipStream_t stream) {
    const int*   edge_index = (const int*)d_in[0];      // [2][E] (int32 under JAX x64-off)
    const float* weight     = (const float*)d_in[1];    // [2E]
    const float* emb        = (const float*)d_in[2];    // [N][256]
    const float* W1_l       = (const float*)d_in[3];
    const float* b1_l       = (const float*)d_in[4];
    const float* W1_r       = (const float*)d_in[5];
    const float* W2_l       = (const float*)d_in[6];
    const float* b2_l       = (const float*)d_in[7];
    const float* W2_r       = (const float*)d_in[8];
    float* out = (float*)d_out;

    const int E = in_sizes[0] / 2;
    const int N = in_sizes[2] / DIM;
    const int n_row_tiles = (N + 15) / 16;

    // workspace layout
    const size_t szX = (size_t)N * DIM * sizeof(float);
    char* ws = (char*)d_ws;
    float*  x    = (float*)(ws);
    float*  x1   = (float*)(ws + szX);
    float*  agg  = (float*)(ws + 2 * szX);
    __bf16* Bt1  = (__bf16*)(ws + 3 * szX);
    __bf16* Bt2  = Bt1 + (size_t)DIM * 2 * DIM;

    // pack weights to bf16 [Wl | Wr]
    k_build_bt<<<(DIM * 2 * DIM + 255) / 256, 256, 0, stream>>>(W1_l, W1_r, Bt1);
    k_build_bt<<<(DIM * 2 * DIM + 255) / 256, 256, 0, stream>>>(W2_l, W2_r, Bt2);

    // x = L2-normalize(emb)
    k_normalize<<<(N + 7) / 8, 256, 0, stream>>>(emb, x, N);

    dim3 ggrid((n_row_tiles + 7) / 8, 4);

    // ---- layer 1 ----
    k_zero<<<2048, 256, 0, stream>>>(agg, (long long)N * (DIM / 4));
    k_scatter<<<(2 * E + 7) / 8, 256, 0, stream>>>(edge_index, weight, x, agg, E);
    k_gemm<<<ggrid, 256, 0, stream>>>(agg, x, Bt1, b1_l, nullptr, nullptr, x1,
                                      n_row_tiles, N);

    // ---- layer 2 (epilogue fuses out = x + x1 + leaky(gemm)) ----
    k_zero<<<2048, 256, 0, stream>>>(agg, (long long)N * (DIM / 4));
    k_scatter<<<(2 * E + 7) / 8, 256, 0, stream>>>(edge_index, weight, x1, agg, E);
    k_gemm<<<ggrid, 256, 0, stream>>>(agg, x1, Bt2, b2_l, x, x1, out,
                                      n_row_tiles, N);
}
